// ElmanRNN_pred_10728828305913
// MI455X (gfx1250) — matrix operand order repaired and checked by reading, not compiled, not run on hardware
//
#include <hip/hip_runtime.h>
#include <hip/hip_bf16.h>
#include <math.h>

typedef __attribute__((ext_vector_type(16))) _Float16 v16h;
typedef __attribute__((ext_vector_type(8)))  _Float16 v8h;
typedef __attribute__((ext_vector_type(4)))  _Float16 v4h;
typedef __attribute__((ext_vector_type(8)))  float    v8f;
typedef __attribute__((ext_vector_type(4)))  unsigned int u32x4;
typedef __attribute__((ext_vector_type(8)))  int      i32x8;
typedef __attribute__((ext_vector_type(4)))  int      i32x4;

#define B_   64
#define T_   512
#define IN_  512
#define H_   2048
#define OUT_ 512

#define RECUR_BLOCKS 64u

// TDM-padded LDS layout for the staged A tile: 8B pad every 1KB
// row pitch = 4096 + 4*8 = 4128 bytes; addr(q) = q + 8*(q>>10)
#define A_LDS_BYTES (16 * 4128 + 256)

#if defined(__HIP_DEVICE_COMPILE__) && __has_builtin(__builtin_amdgcn_tensor_load_to_lds)
#define USE_TDM 1
#else
#define USE_TDM 0
#endif

// ---------------------------------------------------------------------------
// WMMA fragment helpers (wave32, 16x16x32 f16 -> f32)
// A layout (16x32, MxK): lanes 0-15 M=lane, lanes 16-31 M=lane-16.
//   per lane: a[0..7] = A[m][kb..kb+7], a[8..15] = A[m][16+kb..16+kb+7],
//   kb = 8*(lane/16).  => two contiguous 16B loads.
// B is pre-packed so each lane's 16 elements are contiguous (one 32B load).
// ---------------------------------------------------------------------------
__device__ __forceinline__ v16h load_a_frag(const _Float16* __restrict__ A,
                                            size_t lda, int m0, int k0, int lane) {
  const _Float16* row = A + (size_t)(m0 + (lane & 15)) * lda + k0 + ((lane >> 4) << 3);
  v8h lo = *(const v8h*)(row);
  v8h hi = *(const v8h*)(row + 16);
  v16h a;
#pragma unroll
  for (int i = 0; i < 8; ++i) { a[i] = lo[i]; a[8 + i] = hi[i]; }
  return a;
}

__device__ __forceinline__ v16h load_b_frag(const _Float16* __restrict__ Wp,
                                            int ktiles, int nt, int kt, int lane) {
  const _Float16* p = Wp + ((size_t)(nt * ktiles + kt) << 9) + (lane << 4);
  return *(const v16h*)p;  // 32 bytes, aligned
}

// A-fragment from the TDM-padded LDS tile; 4x ds_load_b64 on distinct banks.
__device__ __forceinline__ v16h lds_a_frag(const char* AsB, int kt, int lane) {
  int m = lane & 15;
  size_t q0 = ((size_t)m << 12) + (size_t)((kt << 6) + ((lane >> 4) << 4));
  size_t q1 = q0 + 32;
  size_t a0 = q0 + ((q0 >> 10) << 3);
  size_t a1 = q1 + ((q1 >> 10) << 3);
  v4h p0 = *(const v4h*)(AsB + a0);
  v4h p1 = *(const v4h*)(AsB + a0 + 8);
  v4h p2 = *(const v4h*)(AsB + a1);
  v4h p3 = *(const v4h*)(AsB + a1 + 8);
  v16h a;
#pragma unroll
  for (int i = 0; i < 4; ++i) {
    a[i] = p0[i]; a[4 + i] = p1[i]; a[8 + i] = p2[i]; a[12 + i] = p3[i];
  }
  return a;
}

// 2x2 register-blocked tile: 4 fragment loads feed 4 WMMAs per k-step.
__device__ __forceinline__ void wmma_tile_2x2(
    const _Float16* __restrict__ A, size_t lda, int m0,
    const _Float16* __restrict__ Wp, int ktiles, int nt0,
    v8f& c00, v8f& c01, v8f& c10, v8f& c11, int lane) {
  for (int kt = 0; kt < ktiles; ++kt) {
    if (kt + 1 < ktiles) {  // uniform; prefetch next packed-B tiles
      __builtin_prefetch(Wp + ((size_t)(nt0 * ktiles + kt + 1) << 9) + (lane << 4), 0, 0);
      __builtin_prefetch(Wp + ((size_t)((nt0 + 1) * ktiles + kt + 1) << 9) + (lane << 4), 0, 0);
    }
    v16h a0 = load_a_frag(A, lda, m0,      kt * 32, lane);
    v16h a1 = load_a_frag(A, lda, m0 + 16, kt * 32, lane);
    v16h b0 = load_b_frag(Wp, ktiles, nt0,     kt, lane);
    v16h b1 = load_b_frag(Wp, ktiles, nt0 + 1, kt, lane);
    c00 = __builtin_amdgcn_wmma_f32_16x16x32_f16(false, a0, false, b0, (short)0, c00, false, false);
    c01 = __builtin_amdgcn_wmma_f32_16x16x32_f16(false, a0, false, b1, (short)0, c01, false, false);
    c10 = __builtin_amdgcn_wmma_f32_16x16x32_f16(false, a1, false, b0, (short)0, c10, false, false);
    c11 = __builtin_amdgcn_wmma_f32_16x16x32_f16(false, a1, false, b1, (short)0, c11, false, false);
  }
}

// ---------------------------------------------------------------------------
// Prep kernels
// ---------------------------------------------------------------------------
__global__ void k_cvt_f32_to_f16(const float* __restrict__ in,
                                 _Float16* __restrict__ out, size_t n) {
  size_t i = (size_t)blockIdx.x * blockDim.x + threadIdx.x;
  if (i < n) out[i] = (_Float16)in[i];
}

__global__ void k_zero_u32(unsigned* __restrict__ p) { *p = 0u; }

// Pack W[nrow][K] (row-major f32) into WMMA-B tiles:
// packed[(nt*ktiles+kt)*512 + lane*16 + e] = W[nt*16+(lane&15)][kt*32+8*(lane/16)+(e&7)+16*(e>>3)]
__global__ void k_pack_w(const float* __restrict__ W, _Float16* __restrict__ Wp,
                         int K, int ktiles, size_t n) {
  size_t i = (size_t)blockIdx.x * blockDim.x + threadIdx.x;
  if (i >= n) return;
  int e    = (int)(i & 15);
  int lane = (int)((i >> 4) & 31);
  size_t tile = i >> 9;
  int nt = (int)(tile / ktiles);
  int kt = (int)(tile % ktiles);
  int k  = kt * 32 + ((lane >> 4) << 3) + (e & 7) + ((e >> 3) << 4);
  int nc = nt * 16 + (lane & 15);
  Wp[i] = (_Float16)W[(size_t)nc * K + k];
}

// ---------------------------------------------------------------------------
// xp = x @ Wx^T + bx   -> f16, stored in hs buffer layout [B*T][H]
// one wave computes a 32x32 output block (2x2 tiles)
// ---------------------------------------------------------------------------
__global__ void k_xp(const _Float16* __restrict__ x16,
                     const _Float16* __restrict__ Wxp,
                     const float* __restrict__ bx,
                     _Float16* __restrict__ xp) {
  int wave = (int)((blockIdx.x * blockDim.x + threadIdx.x) >> 5);
  int lane = (int)(threadIdx.x & 31);
  int nt0 = (wave & 63) << 1;  // 64 n-groups of 2 tiles
  int mt0 = (wave >> 6) << 1;  // 1024 m-groups of 2 tiles
  v8f c00 = {}, c01 = {}, c10 = {}, c11 = {};
  wmma_tile_2x2(x16, IN_, mt0 * 16, Wxp, IN_ / 32, nt0, c00, c01, c10, c11, lane);
  int ncol = lane & 15;
  int mb   = ((lane >> 4) << 3);
#pragma unroll
  for (int r = 0; r < 8; ++r) {
    int n0 = nt0 * 16 + ncol, n1 = (nt0 + 1) * 16 + ncol;
    size_t r0 = (size_t)(mt0 * 16 + mb + r) * H_;
    size_t r1 = (size_t)((mt0 + 1) * 16 + mb + r) * H_;
    xp[r0 + n0] = (_Float16)(c00[r] + bx[n0]);
    xp[r0 + n1] = (_Float16)(c01[r] + bx[n1]);
    xp[r1 + n0] = (_Float16)(c10[r] + bx[n0]);
    xp[r1 + n1] = (_Float16)(c11[r] + bx[n1]);
  }
}

// ---------------------------------------------------------------------------
// Persistent recurrence kernel: all T steps inside one launch, separated by
// an atomic-counter grid barrier.  64 blocks x 8 waves; each wave owns one
// 16x16 tile of h.  All 8 waves of a block share the same 16 rows of h_{t-1};
// those rows are staged into LDS once per step by the Tensor Data Mover
// (tensor_load_to_lds + s_wait_tensorcnt), with TDM's pad feature inserting
// 8B per 1KB so fragment ds_loads avoid bank conflicts.
// ---------------------------------------------------------------------------
__device__ __forceinline__ void grid_barrier(unsigned* __restrict__ bar,
                                             unsigned target) {
  __syncthreads();
  if (threadIdx.x == 0) {
    __threadfence();              // release h_t stores to device scope
    atomicAdd(bar, 1u);
    while (__atomic_load_n(bar, __ATOMIC_RELAXED) < target)
      __builtin_amdgcn_s_sleep(1);
  }
  __syncthreads();
  __builtin_amdgcn_fence(__ATOMIC_ACQUIRE, "agent");  // see other blocks' h_t
}

__global__ void __launch_bounds__(256, 1) k_recur(
    const _Float16* __restrict__ h016,
    const _Float16* __restrict__ Whp,
    const float* __restrict__ bh,
    _Float16* __restrict__ hs,
    unsigned* __restrict__ bar) {
  extern __shared__ char AsBytes[];            // padded A tile (TDM layout)
  const int tid  = (int)threadIdx.x;
  const int lane = tid & 31;
  const int wv   = tid >> 5;
  const int id   = (int)blockIdx.x * 8 + wv;   // 0..511
  const int nt   = id & 127;
  const int mt   = id >> 7;                    // uniform within a block
  const int n    = nt * 16 + (lane & 15);
  const int mbase = mt * 16 + ((lane >> 4) << 3);
  const float bias = bh[n];
  const size_t ldio = (size_t)T_ * H_;
  const int KT = H_ / 32;

  for (int t = 0; t < T_; ++t) {
    const _Float16* hprev = (t == 0) ? h016 : (hs + (size_t)(t - 1) * H_);
    const size_t lda = (t == 0) ? (size_t)H_ : ldio;
    const _Float16* src = hprev + (size_t)(mt * 16) * lda;

#if USE_TDM
    if (wv == 0) {
      // ---- Tensor DMA: 2D tile 2048x16 f16, global(stride lda) -> LDS ----
      unsigned long long ga = (unsigned long long)(uintptr_t)src;
      u32x4 g0;
      g0[0] = 1u;                                   // count=1, user D#
      g0[1] = (unsigned)(uintptr_t)AsBytes;         // lds_addr (bytes)
      g0[2] = (unsigned)ga;                         // global_addr[31:0]
      g0[3] = (unsigned)((ga >> 32) & 0x01FFFFFFull) | (2u << 30);  // addr hi | type=2
      i32x8 g1;
      g1[0] = (int)((1u << 16)      // data_size = 2 bytes
                  | (1u << 20)      // pad_enable
                  | (7u << 22)      // pad_interval = 256 dwords (1KB)
                  | (1u << 25));    // pad_amount   = 2 dwords (8B)
      g1[1] = (int)(((unsigned)H_ & 0xFFFFu) << 16);  // tensor_dim0 = 2048 (lo16)
      g1[2] = (int)(16u << 16);                       // tensor_dim1 = 16
      g1[3] = (int)((unsigned)H_ << 16);              // tile_dim0 = 2048
      g1[4] = 16;                                     // tile_dim1 = 16
      g1[5] = (int)(unsigned)lda;                     // tensor_dim0_stride (elems)
      g1[6] = 0;
      g1[7] = 0;
      i32x4 gz4 = {0, 0, 0, 0};                       // 2D: groups 2/3 unused
      i32x8 gz8 = {0, 0, 0, 0, 0, 0, 0, 0};
      __builtin_amdgcn_tensor_load_to_lds(g0, g1, gz4, gz4, gz8, 0);
      __builtin_amdgcn_s_wait_tensorcnt(0);
    }
#else
    {
      // Fallback: manual global->LDS copy into the identical padded layout.
      const char* srcB = (const char*)src;
      for (int i = tid; i < 4096; i += 256) {        // 4096 x 16B chunks
        int r = i >> 8;
        int c = (i & 255) << 4;
        v8h d = *(const v8h*)(srcB + (size_t)r * (lda * 2) + c);
        size_t q = ((size_t)r << 12) + (size_t)c;
        char* dst = AsBytes + q + ((q >> 10) << 3);
        v4h d0, d1;
#pragma unroll
        for (int j = 0; j < 4; ++j) { d0[j] = d[j]; d1[j] = d[4 + j]; }
        *(v4h*)dst = d0;
        *(v4h*)(dst + 8) = d1;
      }
    }
#endif
    __syncthreads();

    _Float16* io = hs + (size_t)t * H_;  // xp_t in, h_t out (in-place)
    v8f acc;
#pragma unroll
    for (int r = 0; r < 8; ++r)
      acc[r] = (float)io[(size_t)(mbase + r) * ldio + n];

    for (int kt = 0; kt < KT; ++kt) {
      if (kt + 1 < KT)
        __builtin_prefetch(Whp + ((size_t)(nt * KT + kt + 1) << 9) + (lane << 4), 0, 0);
      v16h a = lds_a_frag(AsBytes, kt, lane);
      v16h b = load_b_frag(Whp, KT, nt, kt, lane);
      acc = __builtin_amdgcn_wmma_f32_16x16x32_f16(false, a, false, b,
                                                   (short)0, acc, false, false);
    }
#pragma unroll
    for (int r = 0; r < 8; ++r)
      io[(size_t)(mbase + r) * ldio + n] = (_Float16)tanhf(acc[r] + bias);

    grid_barrier(bar, (unsigned)(t + 1) * RECUR_BLOCKS);
  }
}

// ---------------------------------------------------------------------------
// logits = hs @ Wo^T + bo  -> f32 [B*T][OUT];  2x2 tiles per wave
// ---------------------------------------------------------------------------
__global__ void k_logits(const _Float16* __restrict__ hs,
                         const _Float16* __restrict__ Wop,
                         const float* __restrict__ bo,
                         float* __restrict__ logits) {
  int wave = (int)((blockIdx.x * blockDim.x + threadIdx.x) >> 5);
  int lane = (int)(threadIdx.x & 31);
  int nt0 = (wave & 15) << 1;  // 16 n-groups
  int mt0 = (wave >> 4) << 1;  // 1024 m-groups
  v8f c00 = {}, c01 = {}, c10 = {}, c11 = {};
  wmma_tile_2x2(hs, H_, mt0 * 16, Wop, H_ / 32, nt0, c00, c01, c10, c11, lane);
  int ncol = lane & 15;
  int mb   = ((lane >> 4) << 3);
#pragma unroll
  for (int r = 0; r < 8; ++r) {
    int n0 = nt0 * 16 + ncol, n1 = (nt0 + 1) * 16 + ncol;
    size_t r0 = (size_t)(mt0 * 16 + mb + r) * OUT_;
    size_t r1 = (size_t)((mt0 + 1) * 16 + mb + r) * OUT_;
    logits[r0 + n0] = c00[r] + bo[n0];
    logits[r0 + n1] = c01[r] + bo[n1];
    logits[r1 + n0] = c10[r] + bo[n0];
    logits[r1 + n1] = c11[r] + bo[n1];
  }
}

// ---------------------------------------------------------------------------
// softmax over OUT per row; one block (256 threads) per row, 2 elems/thread
// ---------------------------------------------------------------------------
__global__ void k_softmax(const float* __restrict__ logits, float* __restrict__ out) {
  __shared__ float red[256];
  size_t row = blockIdx.x;
  const float* L = logits + row * OUT_;
  int t = threadIdx.x;
  float v0 = L[t], v1 = L[t + 256];
  red[t] = fmaxf(v0, v1);
  __syncthreads();
  for (int s = 128; s > 0; s >>= 1) {
    if (t < s) red[t] = fmaxf(red[t], red[t + s]);
    __syncthreads();
  }
  float m = red[0];
  __syncthreads();
  float e0 = __expf(v0 - m), e1 = __expf(v1 - m);
  red[t] = e0 + e1;
  __syncthreads();
  for (int s = 128; s > 0; s >>= 1) {
    if (t < s) red[t] += red[t + s];
    __syncthreads();
  }
  float inv = 1.0f / red[0];
  out[row * OUT_ + t]       = e0 * inv;
  out[row * OUT_ + t + 256] = e1 * inv;
}

// ---------------------------------------------------------------------------
// Host launcher.  inputs: 0:x 1:h0 2:Wx 3:bx 4:Wh 5:bh 6:Wo 7:bo
// ---------------------------------------------------------------------------
extern "C" void kernel_launch(void* const* d_in, const int* in_sizes, int n_in,
                              void* d_out, int out_size, void* d_ws, size_t ws_size,
                              hipStream_t stream) {
  const float* x  = (const float*)d_in[0];
  const float* h0 = (const float*)d_in[1];
  const float* Wx = (const float*)d_in[2];
  const float* bx = (const float*)d_in[3];
  const float* Wh = (const float*)d_in[4];
  const float* bh = (const float*)d_in[5];
  const float* Wo = (const float*)d_in[6];
  const float* bo = (const float*)d_in[7];
  float* out = (float*)d_out;

  // Workspace carve-up (all naturally 256B-aligned)
  char* p = (char*)d_ws;
  _Float16* x16  = (_Float16*)p;  p += (size_t)B_ * T_ * IN_ * 2;   // 32 MB
  _Float16* Wxp  = (_Float16*)p;  p += (size_t)H_ * IN_ * 2;        // 2 MB
  _Float16* Whp  = (_Float16*)p;  p += (size_t)H_ * H_ * 2;         // 8 MB
  _Float16* Wop  = (_Float16*)p;  p += (size_t)OUT_ * H_ * 2;       // 2 MB
  _Float16* h016 = (_Float16*)p;  p += (size_t)B_ * H_ * 2;         // 256 KB
  _Float16* hs   = (_Float16*)p;  p += (size_t)B_ * T_ * H_ * 2;    // 128 MB (xp->hs in-place)
  float*    lg   = (float*)p;     p += (size_t)B_ * T_ * OUT_ * 4;  // 64 MB
  unsigned* bar  = (unsigned*)p;  p += 256;

  // --- prep: casts + weight packing + barrier reset ---
  k_zero_u32<<<1, 1, 0, stream>>>(bar);
  {
    size_t n = (size_t)B_ * T_ * IN_;
    k_cvt_f32_to_f16<<<(unsigned)((n + 255) / 256), 256, 0, stream>>>(x, x16, n);
  }
  {
    size_t n = (size_t)B_ * H_;
    k_cvt_f32_to_f16<<<(unsigned)((n + 255) / 256), 256, 0, stream>>>(h0, h016, n);
  }
  {
    size_t n = (size_t)H_ * IN_;  // Wx: [H][IN], K=IN
    k_pack_w<<<(unsigned)((n + 255) / 256), 256, 0, stream>>>(Wx, Wxp, IN_, IN_ / 32, n);
  }
  {
    size_t n = (size_t)H_ * H_;   // Wh: [H][H], K=H
    k_pack_w<<<(unsigned)((n + 255) / 256), 256, 0, stream>>>(Wh, Whp, H_, H_ / 32, n);
  }
  {
    size_t n = (size_t)OUT_ * H_; // Wo: [OUT][H], K=H
    k_pack_w<<<(unsigned)((n + 255) / 256), 256, 0, stream>>>(Wo, Wop, H_, H_ / 32, n);
  }

  // --- xp = x @ Wx^T + bx  (written into hs buffer) ---
  {
    unsigned waves = (B_ * T_ / 32) * (H_ / 32);  // 65536 (2x2 tiles per wave)
    k_xp<<<waves / 8, 256, 0, stream>>>(x16, Wxp, bx, hs);
  }

  // --- recurrence: single persistent kernel, TDM-staged A, grid barrier ---
  k_recur<<<RECUR_BLOCKS, 256, A_LDS_BYTES, stream>>>(h016, Whp, bh, hs, bar);

  // --- readout + softmax ---
  {
    unsigned waves = (B_ * T_ / 32) * (OUT_ / 32);  // 16384 (2x2 tiles per wave)
    k_logits<<<waves / 8, 256, 0, stream>>>(hs, Wop, bo, lg);
  }
  k_softmax<<<B_ * T_, 256, 0, stream>>>(lg, out);
}
